// TGAT_pr_41961830482202
// MI455X (gfx1250) — matrix operand order repaired and checked by reading, not compiled, not run on hardware
//
#include <hip/hip_runtime.h>

#define NN 50000
#define NE 800000
#define DIM 128
#define QLD 512      // qkvs row stride: [q|k|v|s] each 128
#define NG 64
#define DDIM 512
#define ODIM 16

typedef __bf16 bf16;
typedef __attribute__((ext_vector_type(16))) __bf16 v16bf;
typedef __attribute__((ext_vector_type(8)))  __bf16 v8bf;
typedef __attribute__((ext_vector_type(8)))  float  v8f;

static __device__ __forceinline__ void atomicMaxF(float* a, float v) {
  if (v >= 0.f) atomicMax((int*)a, __float_as_int(v));
  else          atomicMin((unsigned int*)a, __float_as_uint(v));
}

// ---------------- elementwise utility ----------------
__global__ void k_fill(float* p, float v, int n) {
  int i = blockIdx.x * blockDim.x + threadIdx.x;
  if (i < n) p[i] = v;
}
__global__ void k_copyf(float* dst, const float* src, int n) {
  int i = blockIdx.x * blockDim.x + threadIdx.x;
  if (i < n) dst[i] = src[i];
}
__global__ void k_f2bf(bf16* dst, const float* src, int n) {
  int i = blockIdx.x * blockDim.x + threadIdx.x;
  if (i < n) dst[i] = (bf16)src[i];
}

// ---------------- temporal encoding -> bf16 activations ----------------
__global__ void k_temporal(const float* __restrict__ x, const float* __restrict__ t,
                           bf16* __restrict__ hbf) {
  int i = blockIdx.x * blockDim.x + threadIdx.x;
  if (i >= NN * DIM) return;
  int node = i >> 7, d = i & 127;
  float ang = t[node] * __expf(-0.07195578f * (float)(d & ~1));  // exp(-(d&~1)*ln(1e4)/128)
  float pe = (d & 1) ? __cosf(ang) : __sinf(ang);
  hbf[i] = (bf16)(x[i] + pe);
}

// ---------------- pack f32 weights [K,N] into WMMA B-tile bf16 layout ----------------
// tile (kt,nt): per lane l: column n = nt*16 + (l&15), K = kt*32 + ((l>=16)?16:0) + j
__global__ void k_packW(const float* __restrict__ W, bf16* __restrict__ dst,
                        int K, int N, int ntOff, int ntTot) {
  int i = blockIdx.x * blockDim.x + threadIdx.x;
  if (i >= K * N) return;
  int k = i / N, n = i - k * N;
  int kt = k >> 5, krem = k & 31;
  int lane = ((krem & 16) ? 16 : 0) + (n & 15);
  int j = krem & 15;
  int nt = ntOff + (n >> 4);
  dst[((size_t)(kt * ntTot + nt) * 32 + lane) * 16 + j] = (bf16)W[i];
}

// ---------------- WMMA GEMM, 1 Mtile x 4 Ntiles per wave ----------------
// C[M, 16*nTilesN] = relu?( A[M,K](bf16) * Bp + bias );  nTilesN % 4 == 0
template <int K>
__global__ void k_gemm_wmma(const bf16* __restrict__ A, const bf16* __restrict__ Bp,
                            const float* __restrict__ bias, float* __restrict__ C,
                            int mTiles, int nTilesN, int doRelu) {
  int wave = blockIdx.x * (blockDim.x >> 5) + (threadIdx.x >> 5);
  int lane = threadIdx.x & 31;
  int nGroups = nTilesN >> 2;
  if (wave >= mTiles * nGroups) return;       // wave-uniform: EXEC all-1s inside
  int mt = wave / nGroups;
  int ng = wave - mt * nGroups;
  int half = lane >> 4;                        // 0: lanes 0-15, 1: lanes 16-31
  int l16  = lane & 15;

  const bf16* arow  = A + (size_t)(mt * 16 + l16) * K + half * 16;      // contiguous k order (16h + s), same as Bp
  const bf16* bbase = Bp + ((size_t)(ng * 4) * 32 + lane) * 16;         // 4 adjacent N-tiles

  v8f acc[4];
#pragma unroll
  for (int j = 0; j < 4; ++j) {
    float bv = bias[(ng * 4 + j) * 16 + l16];
#pragma unroll
    for (int r = 0; r < 8; ++r) acc[j][r] = bv;
  }

#pragma unroll
  for (int kt = 0; kt < K / 32; ++kt) {
    union { v16bf v; v8bf h[2]; } a;
    const v8bf* ap = (const v8bf*)(arow + kt * 32);
    a.h[0] = ap[0];   // K = kt*32 + half*16 + 0..7
    a.h[1] = ap[1];   // K = kt*32 + half*16 + 8..15
    const v16bf* bp = (const v16bf*)(bbase + (size_t)kt * nTilesN * 512);
#pragma unroll
    for (int j = 0; j < 4; ++j) {
      v16bf b = bp[j * 32];     // tile nt = ng*4+j at this kt (512 elems = 32 v16bf apart)
      acc[j] = __builtin_amdgcn_wmma_f32_16x16x32_bf16(false, a.v, false, b,
                                                       (short)0, acc[j], false, false);
    }
  }
  int ldc = nTilesN * 16;
  float* crow = C + (size_t)(mt * 16 + half * 8) * ldc + l16;
#pragma unroll
  for (int j = 0; j < 4; ++j)
#pragma unroll
    for (int r = 0; r < 8; ++r) {
      float v = acc[j][r];
      if (doRelu) v = fmaxf(v, 0.f);
      crow[(size_t)r * ldc + (ng * 4 + j) * 16] = v;
    }
}

// ---------------- edge attention ----------------
__global__ void k_edge_logits(const int* __restrict__ ei, const float* __restrict__ qkvs,
                              float* __restrict__ logits, float* __restrict__ maxb) {
  int i = blockIdx.x * blockDim.x + threadIdx.x;
  if (i >= NE * 2) return;
  int e = i >> 1, h = i & 1;
  int src = ei[e], dst = ei[NE + e];
  const float4* qp = (const float4*)(qkvs + (size_t)dst * QLD + h * 64);
  const float4* kp = (const float4*)(qkvs + (size_t)src * QLD + 128 + h * 64);
  float s = 0.f;
#pragma unroll
  for (int c = 0; c < 16; ++c) {
    float4 q = qp[c], k = kp[c];
    s += q.x * k.x + q.y * k.y + q.z * k.z + q.w * k.w;
  }
  s *= 0.125f;  // 1/sqrt(64)
  logits[i] = s;
  atomicMaxF(&maxb[dst * 2 + h], s);
}

__global__ void k_edge_exp(const int* __restrict__ ei, float* __restrict__ logits,
                           const float* __restrict__ maxb, float* __restrict__ denom) {
  int i = blockIdx.x * blockDim.x + threadIdx.x;
  if (i >= NE * 2) return;
  int e = i >> 1, h = i & 1;
  int dst = ei[NE + e];
  float ee = __expf(logits[i] - maxb[dst * 2 + h]);
  logits[i] = ee;
  atomicAdd(&denom[dst * 2 + h], ee);
}

__global__ void k_edge_scatter(const int* __restrict__ ei, const float* __restrict__ qkvs,
                               const float* __restrict__ ebuf, const float* __restrict__ denom,
                               float* __restrict__ accum) {
  int i = blockIdx.x * blockDim.x + threadIdx.x;
  if (i >= NE * DIM) return;
  int e = i >> 7, c = i & 127;
  int h = c >> 6;
  int src = ei[e], dst = ei[NE + e];
  float alpha = ebuf[e * 2 + h] / (denom[dst * 2 + h] + 1e-16f);
  atomicAdd(&accum[(size_t)dst * DIM + c], alpha * qkvs[(size_t)src * QLD + 256 + c]);
}

// ---------------- relu -> BN(eval) -> relu; write f32 (in-place) + bf16 ----------------
__global__ void k_post(const float* __restrict__ qkvs, const float* __restrict__ gamma,
                       const float* __restrict__ beta, const float* __restrict__ mean,
                       const float* __restrict__ var, float* __restrict__ accum,
                       bf16* __restrict__ hbf) {
  int i = blockIdx.x * blockDim.x + threadIdx.x;
  if (i >= NN * DIM) return;
  int n = i >> 7, c = i & 127;
  float a = fmaxf(accum[i] + qkvs[(size_t)n * QLD + 384 + c], 0.f);  // + skip (x@Ws+bs)
  float b = (a - mean[c]) * rsqrtf(var[c] + 1e-5f) * gamma[c] + beta[c];
  float r = fmaxf(b, 0.f);
  accum[i] = r;
  hbf[i] = (bf16)r;
}

// ---------------- pooling ----------------
__global__ void k_pool_sum(const float* __restrict__ h, const int* __restrict__ batch,
                           float* __restrict__ sums) {
  int i = blockIdx.x * blockDim.x + threadIdx.x;
  if (i >= NN * DIM) return;
  int n = i >> 7, c = i & 127;
  atomicAdd(&sums[batch[n] * DIM + c], h[i]);
}
__global__ void k_count(const int* __restrict__ batch, float* __restrict__ cnt) {
  int n = blockIdx.x * blockDim.x + threadIdx.x;
  if (n < NN) atomicAdd(&cnt[batch[n]], 1.f);
}
__global__ void k_pool_div(const float* __restrict__ sums, const float* __restrict__ cnt,
                           float* __restrict__ g) {
  int i = blockIdx.x * blockDim.x + threadIdx.x;
  if (i >= NG * DIM) return;
  g[i] = sums[i] / fmaxf(cnt[i >> 7], 1.f);
}

// ---------------- concat [g | doc_emb] -> bf16 for fusion GEMM ----------------
__global__ void k_concat_bf(const float* __restrict__ g, const float* __restrict__ de,
                            bf16* __restrict__ cat) {
  int i = blockIdx.x * blockDim.x + threadIdx.x;
  if (i >= NG * 256) return;
  int bI = i >> 8, c = i & 255;
  cat[i] = (bf16)((c < 128) ? g[bI * 128 + c] : de[bI * 128 + (c - 128)]);
}

// ---------------- tiny dense heads (task/time) ----------------
__global__ void k_dense(const float* __restrict__ A, const float* __restrict__ W,
                        const float* __restrict__ b, float* __restrict__ C,
                        int rows, int K, int N, int doRelu) {
  int i = blockIdx.x * blockDim.x + threadIdx.x;
  if (i >= rows * N) return;
  int r = i / N, n = i - r * N;
  float s = b[n];
  for (int k = 0; k < K; ++k) s += A[(size_t)r * K + k] * W[(size_t)k * N + n];
  if (doRelu) s = fmaxf(s, 0.f);
  C[i] = s;
}

static inline int nb(long long n, int b) { return (int)((n + b - 1) / b); }

extern "C" void kernel_launch(void* const* d_in, const int* in_sizes, int n_in,
                              void* d_out, int out_size, void* d_ws, size_t ws_size,
                              hipStream_t stream) {
  (void)in_sizes; (void)n_in; (void)out_size; (void)ws_size;
  const float* x     = (const float*)d_in[0];
  const int*   ei    = (const int*)d_in[1];
  const int*   batch = (const int*)d_in[2];
  const float* ts    = (const float*)d_in[3];
  const float* doc   = (const float*)d_in[4];
  const float* Wl[2][4] = {{(const float*)d_in[5], (const float*)d_in[7],
                            (const float*)d_in[9], (const float*)d_in[11]},
                           {(const float*)d_in[13], (const float*)d_in[15],
                            (const float*)d_in[17], (const float*)d_in[19]}};
  const float* bl[2][4] = {{(const float*)d_in[6], (const float*)d_in[8],
                            (const float*)d_in[10], (const float*)d_in[12]},
                           {(const float*)d_in[14], (const float*)d_in[16],
                            (const float*)d_in[18], (const float*)d_in[20]}};
  const float* bn[2][4] = {{(const float*)d_in[21], (const float*)d_in[22],
                            (const float*)d_in[23], (const float*)d_in[24]},
                           {(const float*)d_in[25], (const float*)d_in[26],
                            (const float*)d_in[27], (const float*)d_in[28]}};
  const float *Wdoc = (const float*)d_in[29], *bdoc = (const float*)d_in[30];
  const float *Wfus = (const float*)d_in[31], *bfus = (const float*)d_in[32];
  const float *Wtask = (const float*)d_in[33], *btask = (const float*)d_in[34];
  const float *Wtime = (const float*)d_in[35], *btime = (const float*)d_in[36];
  float* out = (float*)d_out;

  // ---- workspace carve ----
  char* w = (char*)d_ws; size_t off = 0;
  auto carve = [&](size_t bytes) -> void* {
    void* p = w + off; off += (bytes + 255) & ~(size_t)255; return p;
  };
  bf16*  hbf    = (bf16*)carve((size_t)NN * DIM * sizeof(bf16));
  float* qkvs   = (float*)carve((size_t)NN * QLD * 4);
  float* accum  = (float*)carve((size_t)NN * DIM * 4);
  float* ebuf   = (float*)carve((size_t)NE * 2 * 4);   // logits, then exp in place
  float* maxb   = (float*)carve((size_t)NN * 2 * 4);
  float* denom  = (float*)carve((size_t)NN * 2 * 4);
  bf16*  packW  = (bf16*)carve((size_t)DDIM * DIM * sizeof(bf16));  // fits 512x128 max
  float* biasC  = (float*)carve(QLD * 4);
  float* sums   = (float*)carve(NG * DIM * 4);
  float* cnt    = (float*)carve(NG * 4);
  float* gpool  = (float*)carve(NG * DIM * 4);
  float* demb   = (float*)carve(NG * DIM * 4);
  bf16*  doc_bf = (bf16*)carve((size_t)NG * DDIM * sizeof(bf16));
  bf16*  cat_bf = (bf16*)carve((size_t)NG * 256 * sizeof(bf16));
  float* fused  = (float*)carve(NG * DIM * 4);

  const int B = 256;
  const float NEG_INF = -__builtin_inff();

  // ---- temporal encoding -> bf16 ----
  k_temporal<<<nb((long long)NN * DIM, B), B, 0, stream>>>(x, ts, hbf);

  for (int L = 0; L < 2; ++L) {
    // pack [q|k|v|s] weights + biases for fused GEMM (B = [128, 512])
    for (int j = 0; j < 4; ++j) {
      k_packW<<<nb(DIM * DIM, B), B, 0, stream>>>(Wl[L][j], packW, DIM, DIM, j * 8, 32);
      k_copyf<<<1, 128, 0, stream>>>(biasC + j * 128, bl[L][j], 128);
    }
    // fused qkvs GEMM: 3125 Mtiles x 8 Ngroups (4 Ntiles/wave), 8 waves/block
    k_gemm_wmma<DIM><<<nb((long long)3125 * 8, 8), B, 0, stream>>>(
        hbf, packW, biasC, qkvs, 3125, 32, 0);

    // attention (segment softmax over dst)
    k_fill<<<nb(NN * 2, B), B, 0, stream>>>(maxb, NEG_INF, NN * 2);
    k_fill<<<nb(NN * 2, B), B, 0, stream>>>(denom, 0.f, NN * 2);
    k_fill<<<nb((long long)NN * DIM, B), B, 0, stream>>>(accum, 0.f, NN * DIM);
    k_edge_logits<<<nb((long long)NE * 2, B), B, 0, stream>>>(ei, qkvs, ebuf, maxb);
    k_edge_exp<<<nb((long long)NE * 2, B), B, 0, stream>>>(ei, ebuf, maxb, denom);
    k_edge_scatter<<<nb((long long)NE * DIM, B), B, 0, stream>>>(ei, qkvs, ebuf, denom, accum);

    // relu -> BN -> relu ; accum holds f32 result, hbf the bf16 copy for next GEMM
    k_post<<<nb((long long)NN * DIM, B), B, 0, stream>>>(
        qkvs, bn[L][0], bn[L][1], bn[L][2], bn[L][3], accum, hbf);
  }

  // ---- global mean pool ----
  k_fill<<<nb(NG * DIM, B), B, 0, stream>>>(sums, 0.f, NG * DIM);
  k_fill<<<1, NG, 0, stream>>>(cnt, 0.f, NG);
  k_pool_sum<<<nb((long long)NN * DIM, B), B, 0, stream>>>(accum, batch, sums);
  k_count<<<nb(NN, B), B, 0, stream>>>(batch, cnt);
  k_pool_div<<<nb(NG * DIM, B), B, 0, stream>>>(sums, cnt, gpool);

  // ---- doc embedding: relu(doc @ Wdoc + bdoc) via WMMA (K=512) ----
  k_f2bf<<<nb(NG * DDIM, B), B, 0, stream>>>(doc_bf, doc, NG * DDIM);
  k_packW<<<nb(DDIM * DIM, B), B, 0, stream>>>(Wdoc, packW, DDIM, DIM, 0, 8);
  k_copyf<<<1, 128, 0, stream>>>(biasC, bdoc, 128);
  k_gemm_wmma<DDIM><<<1, B, 0, stream>>>(doc_bf, packW, biasC, demb, 4, 8, 1);

  // ---- fusion: relu([g | doc_emb] @ Wfus + bfus) via WMMA (K=256) ----
  k_concat_bf<<<nb(NG * 256, B), B, 0, stream>>>(gpool, demb, cat_bf);
  k_packW<<<nb(256 * DIM, B), B, 0, stream>>>(Wfus, packW, 256, DIM, 0, 8);
  k_copyf<<<1, 128, 0, stream>>>(biasC, bfus, 128);
  k_gemm_wmma<256><<<1, B, 0, stream>>>(cat_bf, packW, biasC, fused, 4, 8, 1);

  // ---- heads ----
  k_dense<<<nb(NG * ODIM, B), B, 0, stream>>>(fused, Wtask, btask, out, NG, DIM, ODIM, 0);
  k_dense<<<1, NG, 0, stream>>>(fused, Wtime, btime, out + NG * ODIM, NG, DIM, 1, 0);
}
